// MultiHeadSelfAttention_60146722013598
// MI455X (gfx1250) — matrix operand order repaired and checked by reading, not compiled, not run on hardware
//
#include <hip/hip_runtime.h>
#include <hip/hip_bf16.h>

// ---------------------------------------------------------------------------
// Problem constants (reference: B=16, H=8, W=128, C=768, 8 heads, hd=96,
// local window 7x11). Hardcoded (deterministic; harness dims are fixed).
// ---------------------------------------------------------------------------
#define BATCH 16
#define GH    8           // grid height
#define GW    128         // grid width
#define NTOK  1024        // GH*GW
#define CDIM  768
#define NHEAD 8
#define HD    96          // head dim
#define ROWS  16384       // BATCH*NTOK
#define SCALE 0.1020620726159658f  // 1/sqrt(96)

typedef __attribute__((ext_vector_type(16))) __bf16 v16bf;
typedef __attribute__((ext_vector_type(8)))  __bf16 v8bf;
typedef __attribute__((ext_vector_type(8)))  float  v8f;

union AF { v16bf v; v8bf h[2]; };   // A-operand fragment: two 16B halves

static __device__ __forceinline__ v8f wmma_bf16(v16bf a, v16bf b, v8f c) {
  // D = A(16x32 bf16) * B(32x16 bf16) + C(16x16 f32)
  return __builtin_amdgcn_wmma_f32_16x16x32_bf16(
      /*neg_a=*/false, a, /*neg_b=*/false, b,
      /*c_mod=*/(short)0, c, /*reuse_a=*/false, /*reuse_b=*/false);
}

static __device__ __forceinline__ unsigned short f2bf(float f) {
  unsigned int u = __float_as_uint(f);
  u += 0x7FFFu + ((u >> 16) & 1u);   // round-to-nearest-even
  return (unsigned short)(u >> 16);
}

// ---------------------------------------------------------------------------
// Stage 0: fp32 -> bf16 conversion (grid-stride)
// ---------------------------------------------------------------------------
__global__ void f32_to_bf16(const float* __restrict__ s,
                            unsigned short* __restrict__ d, int n) {
  int i = blockIdx.x * blockDim.x + threadIdx.x;
  int stride = gridDim.x * blockDim.x;
  for (; i < n; i += stride) d[i] = f2bf(s[i]);
}

// ---------------------------------------------------------------------------
// Stage 1: QKV GEMM. out[n,o] = sum_c x[n,c] * w_qkv[o,c]  (o in [0,2304))
// One wave computes a 32(M) x 64(N) tile: 2 A frags x 4 B frags -> 8 WMMAs
// per K-step; every B fragment feeds 2 WMMAs, and all 6 operand loads of a
// K-step are issued before the first WMMA consumes them (latency hiding).
// Epilogue scatters to q[bh,n,d], k[bh,n,d] (row major) and vT[bh,d,n].
// ---------------------------------------------------------------------------
__global__ __launch_bounds__(256) void qkv_gemm(
    const unsigned short* __restrict__ xbf_,
    const unsigned short* __restrict__ wbf_,
    unsigned short* __restrict__ qbf,
    unsigned short* __restrict__ kbf,
    unsigned short* __restrict__ vtbf) {
  const __bf16* X  = (const __bf16*)xbf_;
  const __bf16* Wt = (const __bf16*)wbf_;
  const int lane = threadIdx.x & 31;
  const int wave = blockIdx.x * 8 + (threadIdx.x >> 5);
  const int mtile = wave / 36;          // 512 M tiles of 32 rows
  const int nchnk = wave % 36;          // 36 N chunks of 64
  const int m0 = mtile * 32, o0 = nchnk * 64;
  const int hl = lane >> 4, l15 = lane & 15;

  v8f acc[2][4];
  #pragma unroll
  for (int g = 0; g < 2; ++g)
    #pragma unroll
    for (int j = 0; j < 4; ++j)
      #pragma unroll
      for (int e = 0; e < 8; ++e) acc[g][j][e] = 0.0f;

  const __bf16* arow0 = X + (size_t)(m0 + l15) * CDIM;
  const __bf16* arow1 = X + (size_t)(m0 + 16 + l15) * CDIM;
  for (int kk = 0; kk < CDIM; kk += 32) {
    // ---- issue all operand loads for this K-step first ----
    AF a0, a1;
    const __bf16* ap0 = arow0 + kk + hl * 16;  // A layout: lane<16 K=0..15, lane>=16 K=16..31
    const __bf16* ap1 = arow1 + kk + hl * 16;
    a0.h[0] = *(const v8bf*)ap0;
    a0.h[1] = *(const v8bf*)(ap0 + 8);
    a1.h[0] = *(const v8bf*)ap1;
    a1.h[1] = *(const v8bf*)(ap1 + 8);
    v16bf b[4];
    #pragma unroll
    for (int j = 0; j < 4; ++j) {
      const __bf16* bp = Wt + (size_t)(o0 + 16 * j + l15) * CDIM + kk + hl * 16;
      b[j] = *(const v16bf*)bp;                // B layout: lane holds col, 16 contig K
    }
    __builtin_prefetch((const void*)(ap0 + 32), 0, 1);  // next K step
    __builtin_prefetch((const void*)(ap1 + 32), 0, 1);
    // ---- 8 WMMAs: each B fragment reused by both M halves ----
    #pragma unroll
    for (int j = 0; j < 4; ++j) {
      acc[0][j] = wmma_bf16(a0.v, b[j], acc[0][j]);
      acc[1][j] = wmma_bf16(a1.v, b[j], acc[1][j]);
    }
  }

  // Epilogue: C layout element (m,n): m = r + 8*(lane>=16), n = lane&15
  #pragma unroll
  for (int g = 0; g < 2; ++g) {
    #pragma unroll
    for (int j = 0; j < 4; ++j) {
      const int o = o0 + 16 * j + l15;
      const int which = o / CDIM;            // 0=q 1=k 2=v (uniform per j)
      const int within = o % CDIM;
      const int head = within / HD, d = within % HD;
      #pragma unroll
      for (int r = 0; r < 8; ++r) {
        const int mrow = m0 + 16 * g + r + 8 * hl;
        const int bb = mrow >> 10, n = mrow & (NTOK - 1);
        const int bh = bb * NHEAD + head;
        const unsigned short hv = f2bf(acc[g][j][r]);
        if (which == 0)      qbf[(size_t)(bh * NTOK + n) * HD + d] = hv;
        else if (which == 1) kbf[(size_t)(bh * NTOK + n) * HD + d] = hv;
        else                 vtbf[(size_t)(bh * HD + d) * NTOK + n] = hv;
      }
    }
  }
}

// ---------------------------------------------------------------------------
// Stage 2: windowed flash attention. One wave owns a 16-query tile (all in
// grid row h, cols w0..w0+15). Keys: grid rows h-3..h+3, and exactly the <=2
// aligned 32-key chunks covering cols [w0-5, w0+20]. Online softmax; P is
// repacked C-layout -> A-layout through per-wave LDS.
// ---------------------------------------------------------------------------
__global__ __launch_bounds__(128) void attn_kernel(
    const unsigned short* __restrict__ qbf_,
    const unsigned short* __restrict__ kbf_,
    const unsigned short* __restrict__ vtbf_,
    unsigned short* __restrict__ aout) {
  __shared__ __attribute__((aligned(32))) unsigned short plds[4][16 * 32];

  const int lane = threadIdx.x & 31;
  const int wv = threadIdx.x >> 5;
  const int tid = blockIdx.x * 4 + wv;   // 0..8191 query tiles
  const int bh = tid >> 6;               // 64 tiles per (b,head)
  const int t = tid & 63;
  const int h = t >> 3;
  const int w0 = (t & 7) << 4;
  const int hl = lane >> 4, l15 = lane & 15;

  const __bf16* Q  = (const __bf16*)qbf_  + (size_t)bh * NTOK * HD;
  const __bf16* K  = (const __bf16*)kbf_  + (size_t)bh * NTOK * HD;
  const __bf16* VT = (const __bf16*)vtbf_ + (size_t)bh * HD * NTOK;

  // Load the three Q A-fragments once (16 queries x 96 dims)
  AF qa[3];
  {
    const int qrow = h * GW + w0 + l15;
    #pragma unroll
    for (int f = 0; f < 3; ++f) {
      const __bf16* qp = Q + (size_t)qrow * HD + 32 * f + hl * 16;
      qa[f].h[0] = *(const v8bf*)qp;
      qa[f].h[1] = *(const v8bf*)(qp + 8);
    }
  }

  v8f acc[6];
  #pragma unroll
  for (int j = 0; j < 6; ++j)
    #pragma unroll
    for (int e = 0; e < 8; ++e) acc[j][e] = 0.0f;
  float rmax[8], rsum[8];
  #pragma unroll
  for (int r = 0; r < 8; ++r) { rmax[r] = -1e30f; rsum[r] = 0.0f; }

  const int base0 = (w0 - 5) & ~31;      // first aligned 32-chunk touching window

  for (int dh = -3; dh <= 3; ++dh) {
    const int hp = h + dh;
    if (hp < 0 || hp >= GH) continue;
    #pragma unroll
    for (int c = 0; c < 2; ++c) {
      const int kb = base0 + 32 * c;     // 32-aligned => fully inside or outside
      if (kb < 0 || kb >= GW) continue;
      const int keyc = hp * GW + kb;

      // ---- S = Q K^T over this 32-key chunk (two 16-key N tiles, K=96) ----
      v8f s0, s1;
      #pragma unroll
      for (int e = 0; e < 8; ++e) { s0[e] = 0.0f; s1[e] = 0.0f; }
      #pragma unroll
      for (int f = 0; f < 3; ++f) {
        const __bf16* kp0 = K + (size_t)(keyc + l15) * HD + 32 * f + hl * 16;
        v16bf b0 = *(const v16bf*)kp0;
        s0 = wmma_bf16(qa[f].v, b0, s0);
        const __bf16* kp1 = K + (size_t)(keyc + 16 + l15) * HD + 32 * f + hl * 16;
        v16bf b1 = *(const v16bf*)kp1;
        s1 = wmma_bf16(qa[f].v, b1, s1);
      }

      // ---- scale + local-window mask (|dw| <= 5; |dh| <= 3 by loop) ----
      #pragma unroll
      for (int r = 0; r < 8; ++r) {
        const int qw = w0 + r + 8 * hl;
        const int kw0 = kb + l15, kw1 = kw0 + 16;
        s0[r] = (abs(qw - kw0) <= 5) ? s0[r] * SCALE : -1e30f;
        s1[r] = (abs(qw - kw1) <= 5) ? s1[r] * SCALE : -1e30f;
      }

      // ---- online softmax update ----
      float cmax[8];
      #pragma unroll
      for (int r = 0; r < 8; ++r) cmax[r] = fmaxf(s0[r], s1[r]);
      #pragma unroll
      for (int off = 1; off < 16; off <<= 1)
        #pragma unroll
        for (int r = 0; r < 8; ++r)
          cmax[r] = fmaxf(cmax[r], __shfl_xor(cmax[r], off, 32));
      float alpha[8];
      #pragma unroll
      for (int r = 0; r < 8; ++r) {
        const float nm = fmaxf(rmax[r], cmax[r]);
        alpha[r] = __expf(rmax[r] - nm);
        rmax[r] = nm;
      }
      float ps[8];
      #pragma unroll
      for (int r = 0; r < 8; ++r) {
        s0[r] = __expf(s0[r] - rmax[r]);
        s1[r] = __expf(s1[r] - rmax[r]);
        ps[r] = s0[r] + s1[r];
      }
      #pragma unroll
      for (int off = 1; off < 16; off <<= 1)
        #pragma unroll
        for (int r = 0; r < 8; ++r)
          ps[r] += __shfl_xor(ps[r], off, 32);
      #pragma unroll
      for (int r = 0; r < 8; ++r) rsum[r] = rsum[r] * alpha[r] + ps[r];
      #pragma unroll
      for (int j = 0; j < 6; ++j)
        #pragma unroll
        for (int r = 0; r < 8; ++r) acc[j][r] *= alpha[r];

      // ---- repack P (C layout) -> A layout via per-wave LDS ----
      unsigned short* pl = &plds[wv][0];
      #pragma unroll
      for (int r = 0; r < 8; ++r) {
        const int m = r + 8 * hl;
        pl[m * 32 + l15]      = f2bf(s0[r]);
        pl[m * 32 + 16 + l15] = f2bf(s1[r]);
      }
      asm volatile("s_wait_dscnt 0" ::: "memory");
      AF pa;
      const __bf16* pb = (const __bf16*)pl + l15 * 32 + hl * 16;
      pa.h[0] = *(const v8bf*)pb;
      pa.h[1] = *(const v8bf*)(pb + 8);

      // ---- acc += P * V  (6 d-tiles, K = 32 keys; B from vT, contiguous) ----
      #pragma unroll
      for (int j = 0; j < 6; ++j) {
        const int d = 16 * j + l15;
        const __bf16* vp = VT + (size_t)d * NTOK + keyc + hl * 16;
        v16bf vb = *(const v16bf*)vp;
        acc[j] = wmma_bf16(pa.v, vb, acc[j]);
      }
    }
  }

  // ---- epilogue: normalize, store bf16 into [b, n, head*96 + d] ----
  const int b = bh >> 3, head = bh & 7;
  float inv[8];
  #pragma unroll
  for (int r = 0; r < 8; ++r) inv[r] = 1.0f / rsum[r];
  #pragma unroll
  for (int j = 0; j < 6; ++j) {
    const int d = 16 * j + l15;
    #pragma unroll
    for (int r = 0; r < 8; ++r) {
      const int m = r + 8 * hl;
      const int n = h * GW + w0 + m;
      aout[((size_t)(b * NTOK + n)) * CDIM + head * HD + d] = f2bf(acc[j][r] * inv[r]);
    }
  }
}

// ---------------------------------------------------------------------------
// Stage 3: output projection. out[n,o] = sum_c a[n,c]*w_proj[o,c] + b_proj[o]
// Same 32x64 wave tiling as stage 1; fp32 output.
// ---------------------------------------------------------------------------
__global__ __launch_bounds__(256) void proj_gemm(
    const unsigned short* __restrict__ abf_,
    const unsigned short* __restrict__ wbf_,
    const float* __restrict__ bias,
    float* __restrict__ out) {
  const __bf16* A  = (const __bf16*)abf_;
  const __bf16* Wt = (const __bf16*)wbf_;
  const int lane = threadIdx.x & 31;
  const int wave = blockIdx.x * 8 + (threadIdx.x >> 5);
  const int mtile = wave / 12;         // 512 M tiles of 32 rows
  const int nchnk = wave % 12;         // 12 N chunks of 64
  const int m0 = mtile * 32, o0 = nchnk * 64;
  const int hl = lane >> 4, l15 = lane & 15;

  v8f acc[2][4];
  #pragma unroll
  for (int g = 0; g < 2; ++g)
    #pragma unroll
    for (int j = 0; j < 4; ++j)
      #pragma unroll
      for (int e = 0; e < 8; ++e) acc[g][j][e] = 0.0f;

  const __bf16* arow0 = A + (size_t)(m0 + l15) * CDIM;
  const __bf16* arow1 = A + (size_t)(m0 + 16 + l15) * CDIM;
  for (int kk = 0; kk < CDIM; kk += 32) {
    AF a0, a1;
    const __bf16* ap0 = arow0 + kk + hl * 16;
    const __bf16* ap1 = arow1 + kk + hl * 16;
    a0.h[0] = *(const v8bf*)ap0;
    a0.h[1] = *(const v8bf*)(ap0 + 8);
    a1.h[0] = *(const v8bf*)ap1;
    a1.h[1] = *(const v8bf*)(ap1 + 8);
    v16bf b[4];
    #pragma unroll
    for (int j = 0; j < 4; ++j) {
      const __bf16* bp = Wt + (size_t)(o0 + 16 * j + l15) * CDIM + kk + hl * 16;
      b[j] = *(const v16bf*)bp;
    }
    __builtin_prefetch((const void*)(ap0 + 32), 0, 1);
    __builtin_prefetch((const void*)(ap1 + 32), 0, 1);
    #pragma unroll
    for (int j = 0; j < 4; ++j) {
      acc[0][j] = wmma_bf16(a0.v, b[j], acc[0][j]);
      acc[1][j] = wmma_bf16(a1.v, b[j], acc[1][j]);
    }
  }

  #pragma unroll
  for (int g = 0; g < 2; ++g) {
    #pragma unroll
    for (int j = 0; j < 4; ++j) {
      const int o = o0 + 16 * j + l15;
      const float bb = bias[o];
      #pragma unroll
      for (int r = 0; r < 8; ++r) {
        const int mrow = m0 + 16 * g + r + 8 * hl;
        out[(size_t)mrow * CDIM + o] = acc[g][j][r] + bb;
      }
    }
  }
}

// ---------------------------------------------------------------------------
// Host-side launcher
// ---------------------------------------------------------------------------
extern "C" void kernel_launch(void* const* d_in, const int* in_sizes, int n_in,
                              void* d_out, int out_size, void* d_ws, size_t ws_size,
                              hipStream_t stream) {
  (void)in_sizes; (void)n_in; (void)out_size; (void)ws_size;
  const float* x      = (const float*)d_in[0];   // [16,1024,768]
  const float* w_qkv  = (const float*)d_in[1];   // [2304,768]
  const float* w_proj = (const float*)d_in[2];   // [768,768]
  const float* b_proj = (const float*)d_in[3];   // [768]

  const size_t n_x   = (size_t)ROWS * CDIM;          // 12,582,912
  const size_t n_wq  = (size_t)3 * CDIM * CDIM;      // 1,769,472
  const size_t n_wp  = (size_t)CDIM * CDIM;          //   589,824
  const size_t n_qkv = (size_t)BATCH * NHEAD * NTOK * HD;  // 12,582,912 each

  char* ws = (char*)d_ws;
  unsigned short* xbf  = (unsigned short*)ws;                 ws += n_x  * 2;
  unsigned short* wqbf = (unsigned short*)ws;                 ws += n_wq * 2;
  unsigned short* wpbf = (unsigned short*)ws;                 ws += n_wp * 2;
  unsigned short* qbf  = (unsigned short*)ws;                 ws += n_qkv * 2;
  unsigned short* kbf  = (unsigned short*)ws;                 ws += n_qkv * 2;
  unsigned short* vtbf = (unsigned short*)ws;                 ws += n_qkv * 2;
  unsigned short* abf  = (unsigned short*)ws;                 ws += n_x * 2;

  f32_to_bf16<<<2048, 256, 0, stream>>>(x,      xbf,  (int)n_x);
  f32_to_bf16<<<1024, 256, 0, stream>>>(w_qkv,  wqbf, (int)n_wq);
  f32_to_bf16<<<512,  256, 0, stream>>>(w_proj, wpbf, (int)n_wp);

  // 512 M-tiles * 36 N-chunks = 18432 waves / 8 per block
  qkv_gemm<<<2304, 256, 0, stream>>>(xbf, wqbf, qbf, kbf, vtbf);

  // 8192 query tiles / 4 waves per block
  attn_kernel<<<2048, 128, 0, stream>>>(qbf, kbf, vtbf, abf);

  // 512 M-tiles * 12 N-chunks = 6144 waves / 8 per block
  proj_gemm<<<768, 256, 0, stream>>>(abf, wpbf, b_proj, (float*)d_out);
}
